// VectorQuantizer_19353122636559
// MI455X (gfx1250) — matrix operand order repaired and checked by reading, not compiled, not run on hardware
//
#include <hip/hip_runtime.h>

typedef __attribute__((ext_vector_type(8)))  float  v8f;
typedef __attribute__((ext_vector_type(8)))  __bf16 v8bf;
typedef __attribute__((ext_vector_type(16))) __bf16 v16bf;

#define NUM_K   1024
#define DIM     128
#define HW      4096            // 64*64 spatial positions per batch image
#define NPIX    65536           // 16 * 4096 total pixels
#define PITCH   136             // bf16 elems per LDS tile row (128 + 8 pad)
                                // 272B row stride -> lane m shifts 4 banks/row
                                // -> 32-lane ds_load_b128 is conflict-free.

// ---------------------------------------------------------------------------
// Kernel 1 (prep): per codebook row k
//   e2[k]   = sum_d cb[k][d]^2                       (exact fp32 bias)
//   cbh/cbl = bf16 hi/lo split of cb  (cb ~= cbh + cbl, error ~2^-18 rel)
// ---------------------------------------------------------------------------
__global__ void vq_prep_kernel(const float* __restrict__ cb,
                               float* __restrict__ e2,
                               __bf16* __restrict__ cbh,
                               __bf16* __restrict__ cbl) {
    int k = blockIdx.x * blockDim.x + threadIdx.x;
    if (k >= NUM_K) return;
    const float* row = cb + (size_t)k * DIM;
    float s = 0.f;
#pragma unroll 8
    for (int d = 0; d < DIM; ++d) {
        float v = row[d];
        s += v * v;
        __bf16 h = (__bf16)v;                       // RNE
        cbh[(size_t)k * DIM + d] = h;
        cbl[(size_t)k * DIM + d] = (__bf16)(v - (float)h);
    }
    e2[k] = s;
}

// ---------------------------------------------------------------------------
// Kernel 2: fused  argmin_k( e2[k] - 2 x.e_k )  +  straight-through gather.
//
// Split-bf16 score matmul (fp32-quality):  dot' = eh.yh + eh.yl + el.yh with
// y = -2x, f32 accumulation, C initialized to e2[k] -> D is the score.
//
// Block = 8 waves = 128 pixels.  All waves share the per-kt codebook tile,
// staged in LDS (double-duty: 8x less vmem traffic + low-latency operands).
// ---------------------------------------------------------------------------
__global__ __launch_bounds__(256) void vq_main_kernel(
        const float*  __restrict__ latents,
        const float*  __restrict__ cb,      // original fp32 codebook (gather)
        const __bf16* __restrict__ cbh,
        const __bf16* __restrict__ cbl,
        const float*  __restrict__ e2,
        float* __restrict__ out) {
    __shared__ __align__(16) __bf16 ldsh[16 * PITCH];   // hi tile (padded rows)
    __shared__ __align__(16) __bf16 ldsl[16 * PITCH];   // lo tile
    __shared__ __align__(16) float  ldse2[NUM_K];       // all 1024 biases

    const int tid  = threadIdx.x;
    const int lane = tid & 31;
    const int wave = tid >> 5;
    const int m    = lane & 15;   // pixel column (B/D) and codebook row (A)
    const int half = lane >> 4;

    // Stage e2 once: 256 threads x 16B = 4KB.
    ((float4*)ldse2)[tid] = ((const float4*)e2)[tid];

    // Tile-copy slot for this thread: 16B of the hi tile + 16B of the lo tile.
    const int    crow = tid >> 4;                    // 0..15 codebook row
    const int    ccol = tid & 15;                    // 16B chunk within row
    const size_t gsl  = (size_t)crow * DIM + ccol * 8;
    const int    loff = crow * PITCH + ccol * 8;

    const int pix0 = (blockIdx.x * 8 + wave) * 16;   // 16 consecutive pixels
    const int b    = pix0 >> 12;
    const int rem  = pix0 & (HW - 1);

    const float* inbase  = latents + (size_t)b * DIM * HW + rem;
    float*       outbase = out     + (size_t)b * DIM * HW + rem;

    // Preload B tiles: y = -2x split into bf16 hi/lo, packed per the
    // 16-bit 32x16 B layout (lane N = pixel; elems[0..7]=K(8h..), elems[8..15]=K(16+8h..)).
    v16bf yh[4], yl[4];
#pragma unroll
    for (int c = 0; c < 4; ++c) {
#pragma unroll
        for (int j = 0; j < 16; ++j) {
            int d = c * 32 + (j < 8 ? 8 * half + j : 16 + 8 * half + (j - 8));
            float f = -2.0f * inbase[(size_t)d * HW + m];
            __bf16 h = (__bf16)f;
            yh[c][j] = h;
            yl[c][j] = (__bf16)(f - (float)h);
        }
    }

    // Stage tile 0 into LDS.
    {
        uint4 h0 = *(const uint4*)(cbh + gsl);
        uint4 l0 = *(const uint4*)(cbl + gsl);
        *(uint4*)(ldsh + loff) = h0;
        *(uint4*)(ldsl + loff) = l0;
    }
    __syncthreads();

    float bestVal = 3.0e38f;
    int   bestIdx = 0;

    for (int kt = 0; kt < NUM_K / 16; ++kt) {
        // Prefetch next tile into registers now; the global latency hides
        // under this tile's 12 WMMAs (consumed only after the barrier).
        uint4 ph, pl;
        const bool more = (kt < NUM_K / 16 - 1);
        if (more) {
            const size_t g = (size_t)(kt + 1) * 16 * DIM + gsl;
            ph = *(const uint4*)(cbh + g);
            pl = *(const uint4*)(cbl + g);
        }

        // C init = e2 per row (broadcast ds_load across each half's lanes).
        const float* e2t = ldse2 + kt * 16 + 8 * half;
        v8f acc;
#pragma unroll
        for (int j = 0; j < 8; ++j) acc[j] = e2t[j];

        // A 16x32 layout: lane M = m; elems[0..7]=K(8h..), elems[8..15]=K(16+8h..)
        const __bf16* hrow = ldsh + m * PITCH;
        const __bf16* lrow = ldsl + m * PITCH;
#pragma unroll
        for (int c = 0; c < 4; ++c) {
            v8bf h0 = *(const v8bf*)(hrow + c * 32 + 8 * half);
            v8bf h1 = *(const v8bf*)(hrow + c * 32 + 16 + 8 * half);
            v8bf l0 = *(const v8bf*)(lrow + c * 32 + 8 * half);
            v8bf l1 = *(const v8bf*)(lrow + c * 32 + 16 + 8 * half);
            v16bf ah = __builtin_shufflevector(h0, h1,
                0,1,2,3,4,5,6,7,8,9,10,11,12,13,14,15);
            v16bf al = __builtin_shufflevector(l0, l1,
                0,1,2,3,4,5,6,7,8,9,10,11,12,13,14,15);

            acc = __builtin_amdgcn_wmma_f32_16x16x32_bf16(
                      false, ah, false, yh[c], (short)0, acc, false, false);
            acc = __builtin_amdgcn_wmma_f32_16x16x32_bf16(
                      false, ah, false, yl[c], (short)0, acc, false, false);
            acc = __builtin_amdgcn_wmma_f32_16x16x32_bf16(
                      false, al, false, yh[c], (short)0, acc, false, false);
        }

        // Running argmin: ascending k within a lane, strict < keeps earliest.
#pragma unroll
        for (int j = 0; j < 8; ++j) {
            float v    = acc[j];
            int   kidx = kt * 16 + 8 * half + j;
            if (v < bestVal) { bestVal = v; bestIdx = kidx; }
        }

        // Rotate the staged tile (uniform control flow; EXEC stays all-ones).
        if (more) {
            __syncthreads();                 // everyone done reading tile kt
            *(uint4*)(ldsh + loff) = ph;
            *(uint4*)(ldsl + loff) = pl;
            __syncthreads();                 // tile kt+1 visible to all waves
        }
    }

    // Merge the two half-wave candidate sets for each pixel (ties -> lower k).
    float oV = __shfl_xor(bestVal, 16, 32);
    int   oI = __shfl_xor(bestIdx, 16, 32);
    if (oV < bestVal || (oV == bestVal && oI < bestIdx)) {
        bestVal = oV;
        bestIdx = oI;
    }

    // Straight-through output in exact fp32: out = l + (q - l).
    // Lanes 0-15 write channel row 2i, lanes 16-31 row 2i+1 -> 64B row stores.
    const float* qrow = cb + (size_t)bestIdx * DIM;
#pragma unroll 4
    for (int i = 0; i < 64; ++i) {
        int d = 2 * i + half;
        float q = qrow[d];
        float l = inbase[(size_t)d * HW + m];
        outbase[(size_t)d * HW + m] = l + (q - l);
    }
}

// ---------------------------------------------------------------------------
extern "C" void kernel_launch(void* const* d_in, const int* in_sizes, int n_in,
                              void* d_out, int out_size, void* d_ws, size_t ws_size,
                              hipStream_t stream) {
    (void)in_sizes; (void)n_in; (void)out_size; (void)ws_size;
    const float* latents = (const float*)d_in[0];   // (16, 128, 64, 64)
    const float* cb      = (const float*)d_in[1];   // (1024, 128)
    float*       out     = (float*)d_out;           // (16, 128, 64, 64)

    // Workspace layout: [e2: 4KB][cbh: 256KB][cbl: 256KB]
    float*  e2  = (float*)d_ws;
    __bf16* cbh = (__bf16*)((char*)d_ws + 4096);
    __bf16* cbl = (__bf16*)((char*)d_ws + 4096 + (size_t)NUM_K * DIM * 2);

    vq_prep_kernel<<<NUM_K / 256, 256, 0, stream>>>(cb, e2, cbh, cbl);
    vq_main_kernel<<<NPIX / 128, 256, 0, stream>>>(latents, cb, cbh, cbl, e2, out);
}
